// Attention_50130858279353
// MI455X (gfx1250) — matrix operand order repaired and checked by reading, not compiled, not run on hardware
//
#include <hip/hip_runtime.h>
#include <math.h>

#define B_ 8
#define H_ 64
#define W_ 64
#define C_ 256
#define CFG 32
#define N_ (H_ * W_)        // 4096
#define NROWS (B_ * N_)     // 32768
#define KCHUNK 32

typedef __attribute__((ext_vector_type(4)))  unsigned int u32x4;
typedef __attribute__((ext_vector_type(4)))  float        f32x4;
typedef __attribute__((ext_vector_type(8)))  float        v8f;
typedef __attribute__((ext_vector_type(16))) __bf16       v16bf;

union FragAB {
    v16bf v;
    u32x4 q[2];
    unsigned int d[8];
    __bf16 e[16];
    unsigned short u[16];
};

__device__ __forceinline__ unsigned int f32bits(float f) {
    unsigned u; __builtin_memcpy(&u, &f, 4); return u;
}
// truncate-to-bf16 (single shift; compiler folds into d16_hi stores)
__device__ __forceinline__ unsigned short f2bf_u16(float f) {
    return (unsigned short)(f32bits(f) >> 16);
}
// two floats -> packed 2x bf16 dword via one v_perm_b32
__device__ __forceinline__ unsigned int pack2bf(float a, float b) {
    // result bytes: [0]=a.b2 [1]=a.b3 [2]=b.b2 [3]=b.b3
    // v_perm: S1 bytes are sel 0..3, S0 bytes are sel 4..7
    return __builtin_amdgcn_perm(f32bits(b), f32bits(a), 0x07060302u);
}

// async copy pointer types: <4 x i32> in global (AS1) / LDS (AS3)
typedef int v4i_ __attribute__((vector_size(16)));
typedef __attribute__((address_space(1))) v4i_ gv4_t;
typedef __attribute__((address_space(3))) v4i_ lv4_t;

__device__ __forceinline__ void async_copy16(void* lds, const void* g) {
#if __has_builtin(__builtin_amdgcn_global_load_async_to_lds_b128)
    __builtin_amdgcn_global_load_async_to_lds_b128(
        (gv4_t*)(unsigned long long)(g),
        (lv4_t*)(unsigned int)(unsigned long long)(lds),
        0, 0);
#else
    *(u32x4*)lds = *(const u32x4*)g;
#endif
}
__device__ __forceinline__ void async_wait0() {
#if __has_builtin(__builtin_amdgcn_s_wait_asynccnt)
    __builtin_amdgcn_s_wait_asynccnt(0);
#elif __has_builtin(__builtin_amdgcn_global_load_async_to_lds_b128)
    asm volatile("s_wait_asynccnt 0" ::: "memory");
#endif
}

// ---------------------------------------------------------------------------
// Kernel 1: fused 1x1-conv projections  g = x@Wg+bg, f = x@Wf+bf, h = x@Wh+bh
// One wave computes one 16x16 output tile; K=256 in 8 bf16 WMMA steps.
// grid = (NROWS/16, 2 + 2 + C/16) = (2048, 20), block = 32
// ---------------------------------------------------------------------------
__global__ __launch_bounds__(32)
void proj_kernel(const float* __restrict__ x,
                 const float* __restrict__ wf, const float* __restrict__ wg,
                 const float* __restrict__ wh,
                 const float* __restrict__ bfv, const float* __restrict__ bgv,
                 const float* __restrict__ bhv,
                 unsigned short* __restrict__ gbuf,   // [NROWS][CFG] bf16
                 unsigned short* __restrict__ fbuf,   // [NROWS][CFG] bf16
                 unsigned short* __restrict__ hbuf)   // [NROWS][C]   bf16
{
    const int lane = threadIdx.x & 31;
    const int row0 = blockIdx.x * 16;
    const int ct   = blockIdx.y;

    const float* Wt; const float* bias; unsigned short* out; int outC, col0;
    if (ct < 2)      { Wt = wg; bias = bgv; out = gbuf; outC = CFG; col0 = ct * 16; }
    else if (ct < 4) { Wt = wf; bias = bfv; out = fbuf; outC = CFG; col0 = (ct - 2) * 16; }
    else             { Wt = wh; bias = bhv; out = hbuf; outC = C_;  col0 = (ct - 4) * 16; }

    const int hi   = lane >> 4;
    const int l16  = lane & 15;
    const int koff = hi * 8;

    v8f acc = {};
    const float* xrow = x + (size_t)(row0 + l16) * C_;

    #pragma unroll 1
    for (int kk = 0; kk < C_; kk += 32) {
        FragAB a, b;
        // A-frag (16x32 bf16 layout): two contiguous 8-float runs -> b128 loads
        {
            const f32x4 r0 = *(const f32x4*)(xrow + kk + koff);
            const f32x4 r1 = *(const f32x4*)(xrow + kk + koff + 4);
            const f32x4 r2 = *(const f32x4*)(xrow + kk + koff + 16);
            const f32x4 r3 = *(const f32x4*)(xrow + kk + koff + 20);
            a.d[0] = pack2bf(r0.x, r0.y);  a.d[1] = pack2bf(r0.z, r0.w);
            a.d[2] = pack2bf(r1.x, r1.y);  a.d[3] = pack2bf(r1.z, r1.w);
            a.d[4] = pack2bf(r2.x, r2.y);  a.d[5] = pack2bf(r2.z, r2.w);
            a.d[6] = pack2bf(r3.x, r3.y);  a.d[7] = pack2bf(r3.z, r3.w);
        }
        // B-frag (32x16 bf16 layout): strided weight column, packed pairwise
        #pragma unroll
        for (int j = 0; j < 8; ++j) {
            const float w0 = Wt[(size_t)(kk + hi * 8 + 2 * (j & 3) + (j >> 2) * 16)     * outC + col0 + l16];
            const float w1 = Wt[(size_t)(kk + hi * 8 + 2 * (j & 3) + (j >> 2) * 16 + 1) * outC + col0 + l16];
            b.d[j] = pack2bf(w0, w1);
        }
        acc = __builtin_amdgcn_wmma_f32_16x16x32_bf16(false, a.v, false, b.v,
                                                      (short)0, acc, false, false);
    }

    const float bv = bias[col0 + l16];
    #pragma unroll
    for (int v = 0; v < 8; ++v) {
        const int row = row0 + v + hi * 8;            // C/D layout
        out[(size_t)row * outC + col0 + l16] = f2bf_u16(acc[v] + bv);
    }
}

// ---------------------------------------------------------------------------
// Kernel 2: flash attention.  One block = (batch, 16-query tile), 8 waves.
// Wave 0: S = Q@K^T (2 WMMA) + online softmax; all waves: 2 PV WMMAs for
// their 32 output channels.  V chunks staged through LDS with async copies.
// ---------------------------------------------------------------------------
__global__ __launch_bounds__(256)
void attn_kernel(const unsigned short* __restrict__ gbuf,   // Q
                 const unsigned short* __restrict__ fbuf,   // K
                 const unsigned short* __restrict__ hbuf,   // V
                 const float* __restrict__ x,
                 const float* __restrict__ gamma,
                 float* __restrict__ out)
{
    __shared__ unsigned short Vlds[KCHUNK][C_];   // 16 KB
    __shared__ unsigned short Plds[16][KCHUNK];   // 1 KB
    __shared__ float alphaLds[16];
    __shared__ float linvLds[16];

    const int tid  = threadIdx.x;
    const int w    = tid >> 5;
    const int lane = tid & 31;
    const int hi   = lane >> 4;
    const int l16  = lane & 15;
    const int koff = hi * 8;

    const int b  = blockIdx.x >> 8;               // 256 query tiles per batch
    const int qt = blockIdx.x & 255;
    const int q0 = qt * 16;
    const size_t rowbase = (size_t)b * N_;

    FragAB qf;                                    // Q A-frag, K = CFG = 32
    {
        const unsigned short* qrow = gbuf + (rowbase + q0 + l16) * CFG;
        qf.q[0] = *(const u32x4*)(qrow + koff);
        qf.q[1] = *(const u32x4*)(qrow + koff + 16);
    }

    v8f acc0 = {}, acc1 = {};
    float m_run[8], l_run[8];
    #pragma unroll
    for (int v = 0; v < 8; ++v) { m_run[v] = -1e30f; l_run[v] = 0.f; }

    const int c0 = w * 32;

    #pragma unroll 1
    for (int k0 = 0; k0 < N_; k0 += KCHUNK) {
        {   // cooperative V-chunk copy: 32 keys x 256 ch bf16 = 16 KB contiguous
            const unsigned short* src = hbuf + (rowbase + k0) * C_;
            #pragma unroll
            for (int i = 0; i < 4; ++i) {
                const int seg = tid + i * 256;
                async_copy16(((char*)Vlds) + seg * 16, ((const char*)src) + seg * 16);
            }
            if (k0 + KCHUNK < N_)
                __builtin_prefetch(hbuf + (rowbase + k0 + KCHUNK) * C_, 0, 1);
        }

        if (w == 0) {
            FragAB kf0, kf1;                       // K B-frags for 2x16 keys
            const unsigned short* kr0 = fbuf + (rowbase + k0 + l16) * CFG;
            const unsigned short* kr1 = fbuf + (rowbase + k0 + 16 + l16) * CFG;
            kf0.q[0] = *(const u32x4*)(kr0 + hi * 8);
            kf0.q[1] = *(const u32x4*)(kr0 + hi * 8 + 16);
            kf1.q[0] = *(const u32x4*)(kr1 + hi * 8);
            kf1.q[1] = *(const u32x4*)(kr1 + hi * 8 + 16);

            v8f z = {};
            v8f s0 = __builtin_amdgcn_wmma_f32_16x16x32_bf16(false, qf.v, false, kf0.v,
                                                             (short)0, z, false, false);
            v8f s1 = __builtin_amdgcn_wmma_f32_16x16x32_bf16(false, qf.v, false, kf1.v,
                                                             (short)0, z, false, false);
            #pragma unroll
            for (int v = 0; v < 8; ++v) {
                float a0 = s0[v], a1 = s1[v];
                float t = fmaxf(a0, a1);
                #pragma unroll
                for (int off = 1; off < 16; off <<= 1)
                    t = fmaxf(t, __shfl_xor(t, off, 16));
                const float mnew  = fmaxf(m_run[v], t);
                const float alpha = __expf(m_run[v] - mnew);
                const float p0 = __expf(a0 - mnew);
                const float p1 = __expf(a1 - mnew);
                float rs = p0 + p1;
                #pragma unroll
                for (int off = 1; off < 16; off <<= 1)
                    rs += __shfl_xor(rs, off, 16);
                l_run[v] = l_run[v] * alpha + rs;
                m_run[v] = mnew;
                const int row = v + hi * 8;
                Plds[row][l16]      = f2bf_u16(p0);
                Plds[row][16 + l16] = f2bf_u16(p1);
                if (l16 == 0) alphaLds[row] = alpha;
            }
        }

        async_wait0();
        __syncthreads();

        #pragma unroll
        for (int v = 0; v < 8; ++v) {              // online-softmax rescale
            const float al = alphaLds[v + hi * 8];
            acc0[v] *= al;
            acc1[v] *= al;
        }

        FragAB pf;                                 // P A-frag from LDS
        {
            const unsigned short* prow = &Plds[l16][0];
            pf.q[0] = *(const u32x4*)(prow + koff);
            pf.q[1] = *(const u32x4*)(prow + koff + 16);
        }
        FragAB vf0, vf1;                           // V B-frags from LDS
        #pragma unroll
        for (int i = 0; i < 16; ++i) {
            vf0.u[i] = Vlds[hi * 8 + (i & 7) + (i >> 3) * 16][c0 + l16];
            vf1.u[i] = Vlds[hi * 8 + (i & 7) + (i >> 3) * 16][c0 + 16 + l16];
        }
        acc0 = __builtin_amdgcn_wmma_f32_16x16x32_bf16(false, pf.v, false, vf0.v,
                                                       (short)0, acc0, false, false);
        acc1 = __builtin_amdgcn_wmma_f32_16x16x32_bf16(false, pf.v, false, vf1.v,
                                                       (short)0, acc1, false, false);
        __syncthreads();
    }

    if (w == 0 && l16 == 0) {
        #pragma unroll
        for (int v = 0; v < 8; ++v)
            linvLds[v + hi * 8] = 1.0f / l_run[v];
    }
    __syncthreads();

    const float gm = gamma[0];
    #pragma unroll
    for (int v = 0; v < 8; ++v) {
        const int row = v + hi * 8;
        const float li = linvLds[row];
        const size_t base = (rowbase + q0 + row) * (size_t)C_;
        const int ch0 = c0 + l16;
        const int ch1 = c0 + 16 + l16;
        out[base + ch0] = gm * (acc0[v] * li) + x[base + ch0];
        out[base + ch1] = gm * (acc1[v] * li) + x[base + ch1];
    }
}

extern "C" void kernel_launch(void* const* d_in, const int* in_sizes, int n_in,
                              void* d_out, int out_size, void* d_ws, size_t ws_size,
                              hipStream_t stream) {
    (void)in_sizes; (void)n_in; (void)out_size; (void)ws_size;
    const float* x  = (const float*)d_in[0];
    const float* wf = (const float*)d_in[1];
    const float* wg = (const float*)d_in[2];
    const float* wh = (const float*)d_in[3];
    const float* bf = (const float*)d_in[4];
    const float* bg = (const float*)d_in[5];
    const float* bh = (const float*)d_in[6];
    const float* gm = (const float*)d_in[7];

    unsigned short* gbuf = (unsigned short*)d_ws;            // [NROWS][CFG]
    unsigned short* fbuf = gbuf + (size_t)NROWS * CFG;       // [NROWS][CFG]
    unsigned short* hbuf = fbuf + (size_t)NROWS * CFG;       // [NROWS][C]
    // total workspace: (1M + 1M + 8M) bf16 = 20 MB

    dim3 pgrid(NROWS / 16, 4 + C_ / 16);                     // 2048 x 20
    proj_kernel<<<pgrid, 32, 0, stream>>>(x, wf, wg, wh, bf, bg, bh,
                                          gbuf, fbuf, hbuf);

    attn_kernel<<<dim3(B_ * (N_ / 16)), 256, 0, stream>>>(gbuf, fbuf, hbuf,
                                                          x, gm, (float*)d_out);
}
